// MatchingPursuitAutoEncoder_45320494907996
// MI455X (gfx1250) — matrix operand order repaired and checked by reading, compiled, not run on hardware
//
#include <hip/hip_runtime.h>
#include <stdint.h>

#define ACT_DIM   2048
#define DICT_SIZE 32768
#define S_STEPS   32
#define BATCH     1024

typedef __attribute__((ext_vector_type(16))) __bf16 v16bf;
typedef __attribute__((ext_vector_type(8)))  float  v8f;

union Frag32 { uint4 u[2]; v16bf v; };

__device__ __forceinline__ unsigned short f2bf_rne(float f) {
    unsigned int u = __float_as_uint(f);
    u += 0x7FFFu + ((u >> 16) & 1u);          // round to nearest even
    return (unsigned short)(u >> 16);
}
__device__ __forceinline__ float bf2f(unsigned short h) {
    return __uint_as_float(((unsigned int)h) << 16);
}

__device__ __forceinline__ void wait_asynccnt0() {
#if __has_builtin(__builtin_amdgcn_s_wait_asynccnt)
    __builtin_amdgcn_s_wait_asynccnt(0);
#else
    asm volatile("s_wait_asynccnt 0x0" ::: "memory");
#endif
}

// async copy of 16 bytes: global -> LDS (gfx1250 ASYNCcnt path)
__device__ __forceinline__ void async_g2l_b128(unsigned lds_byte_addr,
                                               const void* gaddr) {
    asm volatile("global_load_async_to_lds_b128 %0, %1, off"
                 :: "v"(lds_byte_addr), "v"(gaddr)
                 : "memory");
}

// ---------------------------------------------------------------------------
// One-time: W (fp32, [A][D] row-major) -> Wt (bf16, [D][A]) transposed so that
// WMMA B-fragment loads are two 16B contiguous K pieces per lane.
// ---------------------------------------------------------------------------
__global__ __launch_bounds__(256) void convert_W(const float* __restrict__ W,
                                                 unsigned short* __restrict__ Wt) {
    __shared__ float tile[32][33];
    const int jt = blockIdx.x * 32;           // dict-column tile
    const int kt = blockIdx.y * 32;           // act-row tile
    const int tx = threadIdx.x;               // 0..31
    const int ty = threadIdx.y;               // 0..7
    #pragma unroll
    for (int yy = ty; yy < 32; yy += 8)
        tile[yy][tx] = W[(size_t)(kt + yy) * DICT_SIZE + (jt + tx)];
    __syncthreads();
    #pragma unroll
    for (int yy = ty; yy < 32; yy += 8)
        Wt[(size_t)(jt + yy) * ACT_DIM + (kt + tx)] = f2bf_rne(tile[tx][yy]);
}

// ---------------------------------------------------------------------------
// r0 = x - b_dec (fp32 + bf16 shadow), best[] = 0
// ---------------------------------------------------------------------------
__global__ __launch_bounds__(256) void init_resid(const float* __restrict__ x,
                                                  const float* __restrict__ b_dec,
                                                  float* __restrict__ r32,
                                                  unsigned short* __restrict__ rbf,
                                                  unsigned long long* __restrict__ best) {
    const int idx = blockIdx.x * 256 + threadIdx.x;
    const int k = idx & (ACT_DIM - 1);
    const float v = x[idx] - b_dec[k];
    r32[idx] = v;
    rbf[idx] = f2bf_rne(v);
    if (idx < BATCH) best[idx] = 0ull;
}

// ---------------------------------------------------------------------------
// Fused corr = r @ W (bf16 WMMA, f32 accum) + per-row abs-argmax.
// Block: 8 waves -> 256 rows x 128 cols. Each wave owns TWO 16-row M-tiles so
// every LDS B-fragment feeds two independent WMMAs (2 matrix ops per DScnt
// wait) and W is swept only BATCH/256 = 4 times per step. W tile staged once
// per block via a double-buffered LDS ping-pong fed by
// global_load_async_to_lds_b128 running one K-step ahead; both A fragments
// are register double-buffered one K-step ahead.
// ---------------------------------------------------------------------------
__global__ __launch_bounds__(256) void corr_argmax(
    const unsigned short* __restrict__ rbf,   // [BATCH][ACT_DIM] bf16
    const unsigned short* __restrict__ Wt,    // [DICT_SIZE][ACT_DIM] bf16
    unsigned long long* __restrict__ best)    // [BATCH]
{
    // [buf][col][k]: per column 32 bf16 (64 B); 8 KB per buffer
    __shared__ unsigned short btile[2][128 * 32];

    const int tid  = threadIdx.x;
    const int lane = tid & 31;
    const int wave = tid >> 5;
    const int m16  = lane & 15;
    const int h    = lane >> 4;
    const int colBase = blockIdx.x * 128;
    const int rowBase = blockIdx.y * 256 + wave * 32;  // wave: rows [rowBase, rowBase+32)

    const unsigned short* arow0 = rbf + (size_t)(rowBase + m16) * ACT_DIM;
    const unsigned short* arow1 = rbf + (size_t)(rowBase + 16 + m16) * ACT_DIM;

    // Producer: 512 16-byte chunks per K-step; thread owns chunks {tid, tid+256}.
    // chunk c -> col = c>>2, piece = c&3 (piece = 8 bf16 along K)
    const int c0col = tid >> 2,        c0p = tid & 3;
    const int c1col = c0col + 64,      c1p = c0p;
    const unsigned short* g0 = Wt + (size_t)(colBase + c0col) * ACT_DIM + c0p * 8;
    const unsigned short* g1 = Wt + (size_t)(colBase + c1col) * ACT_DIM + c1p * 8;
    const unsigned l0 = (unsigned)(uintptr_t)&btile[0][c0col * 32 + c0p * 8];
    const unsigned l1 = (unsigned)(uintptr_t)&btile[0][c1col * 32 + c1p * 8];
    const unsigned BUFB = 128 * 32 * 2;       // 8192 B per buffer

    v8f acc0[8], acc1[8];
    #pragma unroll
    for (int t = 0; t < 8; ++t) {
        acc0[t] = (v8f){0.f,0.f,0.f,0.f,0.f,0.f,0.f,0.f};
        acc1[t] = (v8f){0.f,0.f,0.f,0.f,0.f,0.f,0.f,0.f};
    }

    // prologue: async-fill buffer 0 (K-step 0) and pre-load both A fragments
    async_g2l_b128(l0, g0);
    async_g2l_b128(l1, g1);
    Frag32 afCur0, afCur1;
    afCur0.u[0] = *(const uint4*)(arow0 + 8 * h);
    afCur0.u[1] = *(const uint4*)(arow0 + 16 + 8 * h);
    afCur1.u[0] = *(const uint4*)(arow1 + 8 * h);
    afCur1.u[1] = *(const uint4*)(arow1 + 16 + 8 * h);

    int buf = 0;
    for (int kb = 0; kb < ACT_DIM; kb += 32, buf ^= 1) {
        wait_asynccnt0();                     // this wave's async stores landed
        __syncthreads();                      // everyone's landed; prior-buffer
                                              // readers drained via WMMA deps

        Frag32 afNext0 = afCur0, afNext1 = afCur1;
        if (kb + 32 < ACT_DIM) {
            // stay one K-step ahead on both streams
            const unsigned nb = (buf ^ 1) * BUFB;
            async_g2l_b128(l0 + nb, g0 + kb + 32);
            async_g2l_b128(l1 + nb, g1 + kb + 32);
            afNext0.u[0] = *(const uint4*)(arow0 + kb + 32 + 8 * h);
            afNext0.u[1] = *(const uint4*)(arow0 + kb + 48 + 8 * h);
            afNext1.u[0] = *(const uint4*)(arow1 + kb + 32 + 8 * h);
            afNext1.u[1] = *(const uint4*)(arow1 + kb + 48 + 8 * h);
        }

        // B fragments from LDS: lane = col m16, 16 bytes at k=8h and 16 bytes at k=16+8h.
        // Each fragment feeds two independent WMMAs (both M-tiles).
        const uint4* lb = (const uint4*)&btile[buf][0];
        #pragma unroll
        for (int t = 0; t < 8; ++t) {
            const int col = 16 * t + m16;
            Frag32 bf_;
            bf_.u[0] = lb[col * 4 + h];
            bf_.u[1] = lb[col * 4 + 2 + h];
            acc0[t] = __builtin_amdgcn_wmma_f32_16x16x32_bf16(
                false, afCur0.v, false, bf_.v, (short)0, acc0[t], false, false);
            acc1[t] = __builtin_amdgcn_wmma_f32_16x16x32_bf16(
                false, afCur1.v, false, bf_.v, (short)0, acc1[t], false, false);
        }
        afCur0 = afNext0;
        afCur1 = afNext1;
    }

    // Epilogue: acc*[t][v] is (row = rowBase + rowOff + v + 8h,
    //                          col = colBase + 16t + m16)
    #pragma unroll
    for (int half = 0; half < 2; ++half) {
        const v8f* acc = half ? acc1 : acc0;
        const int rowOff = half ? 16 : 0;
        #pragma unroll
        for (int v = 0; v < 8; ++v) {
            unsigned long long key = 0ull;
            #pragma unroll
            for (int t = 0; t < 8; ++t) {
                const float c = acc[t][v];
                const unsigned int abits = __float_as_uint(fabsf(c));
                const unsigned int col = (unsigned int)(colBase + 16 * t + m16);
                const unsigned long long k2 =
                    ((unsigned long long)abits << 32) | (col << 1) | (c < 0.f ? 1u : 0u);
                key = (k2 > key) ? k2 : key;
            }
            // half-wave butterfly max: lanes 0-15 / 16-31 hold distinct rows
            #pragma unroll
            for (int off = 1; off < 16; off <<= 1) {
                unsigned int hi = (unsigned int)(key >> 32);
                unsigned int lo = (unsigned int)key;
                hi = __shfl_xor(hi, off, 32);
                lo = __shfl_xor(lo, off, 32);
                const unsigned long long o = ((unsigned long long)hi << 32) | lo;
                key = (o > key) ? o : key;
            }
            if (m16 == 0) {
                const int r = rowBase + rowOff + v + 8 * h;
                atomicMax(best + r, key);
            }
        }
    }
}

// ---------------------------------------------------------------------------
// Per-row rank-1 residual update: decode packed winner, r -= z * W[:,j],
// refresh bf16 shadow, reset key slot for the next step.
// ---------------------------------------------------------------------------
__global__ __launch_bounds__(256) void update_resid(
    const unsigned short* __restrict__ Wt,
    float* __restrict__ r32,
    unsigned short* __restrict__ rbf,
    unsigned long long* __restrict__ best)
{
    const int row = blockIdx.x;
    const unsigned long long key = best[row];
    __syncthreads();
    if (threadIdx.x == 0) best[row] = 0ull;

    const unsigned int lo = (unsigned int)key;
    const int col = (int)((lo >> 1) & 0x7FFFu);
    float z = __uint_as_float((unsigned int)(key >> 32));
    if (lo & 1u) z = -z;

    const unsigned short* wcol = Wt + (size_t)col * ACT_DIM;
    float* rrow = r32 + (size_t)row * ACT_DIM;
    unsigned short* brow = rbf + (size_t)row * ACT_DIM;
    #pragma unroll
    for (int k = threadIdx.x; k < ACT_DIM; k += 256) {
        const float v = rrow[k] - z * bf2f(wcol[k]);
        rrow[k] = v;
        brow[k] = f2bf_rne(v);
    }
}

// ---------------------------------------------------------------------------
// x_hat = b_dec + sum z_t W[:,j] == x - r_final  (exact identity)
// ---------------------------------------------------------------------------
__global__ __launch_bounds__(256) void finalize(const float* __restrict__ x,
                                                const float* __restrict__ r32,
                                                float* __restrict__ out) {
    const int idx = blockIdx.x * 256 + threadIdx.x;
    out[idx] = x[idx] - r32[idx];
}

extern "C" void kernel_launch(void* const* d_in, const int* in_sizes, int n_in,
                              void* d_out, int out_size, void* d_ws, size_t ws_size,
                              hipStream_t stream) {
    const float* x     = (const float*)d_in[0];
    const float* W     = (const float*)d_in[1];
    const float* b_dec = (const float*)d_in[2];
    float* out = (float*)d_out;

    char* ws = (char*)d_ws;
    const size_t szWt  = (size_t)ACT_DIM * DICT_SIZE * sizeof(unsigned short); // 128 MB
    const size_t szR32 = (size_t)BATCH * ACT_DIM * sizeof(float);              //   8 MB
    const size_t szRbf = (size_t)BATCH * ACT_DIM * sizeof(unsigned short);     //   4 MB
    unsigned short*     Wt   = (unsigned short*)(ws);
    float*              r32  = (float*)(ws + szWt);
    unsigned short*     rbf  = (unsigned short*)(ws + szWt + szR32);
    unsigned long long* best = (unsigned long long*)(ws + szWt + szR32 + szRbf);

    convert_W<<<dim3(DICT_SIZE / 32, ACT_DIM / 32), dim3(32, 8), 0, stream>>>(W, Wt);
    init_resid<<<(BATCH * ACT_DIM) / 256, 256, 0, stream>>>(x, b_dec, r32, rbf, best);

    for (int s = 0; s < S_STEPS; ++s) {
        corr_argmax<<<dim3(DICT_SIZE / 128, BATCH / 256), 256, 0, stream>>>(rbf, Wt, best);
        update_resid<<<BATCH, 256, 0, stream>>>(Wt, r32, rbf, best);
    }

    finalize<<<(BATCH * ACT_DIM) / 256, 256, 0, stream>>>(x, r32, out);
}
